// Transformer_16363825398223
// MI455X (gfx1250) — matrix operand order repaired and checked by reading, not compiled, not run on hardware
//
#include <hip/hip_runtime.h>
#include <math.h>
#include <stdint.h>

typedef __attribute__((ext_vector_type(16))) __bf16 bf16x16;
typedef __attribute__((ext_vector_type(8)))  __bf16 bf16x8;
typedef __attribute__((ext_vector_type(8)))  float  f32x8;
typedef int v4i __attribute__((vector_size(16)));   // pointee type expected by async-LDS builtins

#define Bz 4
#define Nz 2048
#define Dz 1024
#define Hz 16
#define DHz 64
#define Fz 4096
#define Lz 4
#define BN (Bz*Nz)          // 8192 rows
#define QKVW (3*Dz)         // 3072

#define AS1 __attribute__((address_space(1)))
#define AS3 __attribute__((address_space(3)))

// ---- CDNA5 async-copy feature detection (compile-safe fallbacks) ----
#if defined(__has_builtin)
#  if __has_builtin(__builtin_amdgcn_global_load_async_to_lds_b128)
#    define HAVE_ASYNC_LDS 1
#  else
#    define HAVE_ASYNC_LDS 0
#  endif
#  if __has_builtin(__builtin_amdgcn_s_wait_asynccnt)
#    define HAVE_WAIT_ASYNC 1
#  else
#    define HAVE_WAIT_ASYNC 0
#  endif
#else
#  define HAVE_ASYNC_LDS 0
#  define HAVE_WAIT_ASYNC 0
#endif

template<int Nw>
static __device__ __forceinline__ void wait_async() {
#if HAVE_WAIT_ASYNC
    __builtin_amdgcn_s_wait_asynccnt(Nw);
#else
    asm volatile("s_wait_asynccnt %0" :: "i"(Nw) : "memory");
#endif
}

#if HAVE_ASYNC_LDS
static __device__ __forceinline__ void async_b128(const void* g, void* l) {
    __builtin_amdgcn_global_load_async_to_lds_b128(
        (AS1 v4i*)(AS1 void*)(uintptr_t)g,
        (AS3 v4i*)(AS3 void*)l, 0, 0);
}
#endif

// ---------------- helpers ----------------
static __device__ __forceinline__ bf16x8 ld8(const __bf16* p) {
    return *reinterpret_cast<const bf16x8*>(p);
}
static __device__ __forceinline__ bf16x16 cmb(bf16x8 lo, bf16x8 hi) {
    return __builtin_shufflevector(lo, hi, 0,1,2,3,4,5,6,7,8,9,10,11,12,13,14,15);
}
static __device__ __forceinline__ f32x8 wmma_bf16(bf16x16 a, bf16x16 b, f32x8 c) {
    // (neg_a, A, neg_b, B, c_mod, C, reuse_a, reuse_b)
    return __builtin_amdgcn_wmma_f32_16x16x32_bf16(false, a, false, b, (short)0, c, false, false);
}
static __device__ __forceinline__ float gelu_exact(float v) {
    return 0.5f * v * (1.0f + erff(v * 0.70710678118654752f));
}
// butterfly xor within 16-lane groups via ds_swizzle (group-of-32: {xor<<10 | or<<5 | and})
template<int MASK>
static __device__ __forceinline__ float swz_xor(float v) {
    int i = __builtin_bit_cast(int, v);
    i = __builtin_amdgcn_ds_swizzle(i, (MASK << 10) | 0x1f);
    return __builtin_bit_cast(float, i);
}

// ---------------- weight convert + transpose: W[K][Nn] (f32) -> Wt[Nn][K] (bf16) ----------------
__global__ void convT_kernel(const float* __restrict__ W, __bf16* __restrict__ Wt,
                             int K, int Nn) {
    int idx = blockIdx.x * 256 + threadIdx.x;
    int n = idx / K;
    int k = idx - n * K;
    Wt[idx] = (__bf16)W[(size_t)k * Nn + n];
}

// ---------------- LayerNorm: x f32 [BN x D] -> h bf16, one row per block ----------------
__global__ void ln_kernel(const float* __restrict__ x, const float* __restrict__ g,
                          const float* __restrict__ be, __bf16* __restrict__ out) {
    int row = blockIdx.x;
    const float* xr = x + (size_t)row * Dz;
    float v[4], s = 0.f, s2 = 0.f;
#pragma unroll
    for (int i = 0; i < 4; ++i) {
        v[i] = xr[threadIdx.x + i * 256];
        s += v[i]; s2 += v[i] * v[i];
    }
#pragma unroll
    for (int off = 16; off >= 1; off >>= 1) {
        s  += __shfl_xor(s,  off, 32);
        s2 += __shfl_xor(s2, off, 32);
    }
    __shared__ float rs[8], rs2[8];
    int w = threadIdx.x >> 5, ln = threadIdx.x & 31;
    if (ln == 0) { rs[w] = s; rs2[w] = s2; }
    __syncthreads();
    float ts = 0.f, ts2 = 0.f;
#pragma unroll
    for (int k = 0; k < 8; ++k) { ts += rs[k]; ts2 += rs2[k]; }
    float mu  = ts * (1.0f / Dz);
    float var = ts2 * (1.0f / Dz) - mu * mu;
    float r   = rsqrtf(var + 1e-5f);
#pragma unroll
    for (int i = 0; i < 4; ++i) {
        int c = threadIdx.x + i * 256;
        out[(size_t)row * Dz + c] = (__bf16)((v[i] - mu) * r * g[c] + be[c]);
    }
}

// ---------------- V transpose: qkv bf16 [BN x 3D] -> Vt bf16 [B*H*DH x N] ----------------
__global__ void transpose_v_kernel(const __bf16* __restrict__ qkv, __bf16* __restrict__ Vt) {
    int idx = blockIdx.x * 256 + threadIdx.x;          // over B*H*DH*N
    int n = idx & (Nz - 1);
    int d = (idx >> 11) & (DHz - 1);
    int h = (idx >> 17) & (Hz - 1);
    int b = idx >> 21;
    Vt[idx] = qkv[(size_t)(b * Nz + n) * QKVW + 2 * Dz + h * DHz + d];
}

// ---------------- WMMA GEMM: C[M x Nn] = A[M x K](bf16) * Bt[Nn x K]^T(bf16) ----------------
// MODE 0: outb = C (bf16)   MODE 1: outb = gelu(C + bias) (bf16)   MODE 2: outf += C + bias (f32)
// 256 threads = 8 waves; block tile 128x128, wave tile 32x64, k-step 32.
template<int MODE>
__global__ void gemm_wmma(const __bf16* __restrict__ A, const __bf16* __restrict__ Bt,
                          const float* __restrict__ bias, __bf16* __restrict__ outb,
                          float* __restrict__ outf, int Nn, int K) {
    int lane = threadIdx.x & 31, wave = threadIdx.x >> 5;
    int wm = wave & 3, wn = wave >> 2;
    int l16 = lane & 15, half = lane >> 4;
    int bm0 = blockIdx.y * 128, bn0 = blockIdx.x * 128;
    int bm = bm0 + wm * 32,     bn = bn0 + wn * 64;

    f32x8 acc[2][4] = {};

#if HAVE_ASYNC_LDS
    // LDS staging tiles, 40-element pitch: 80B rows (16B aligned, conflict-free frag reads)
    __shared__ __align__(16) __bf16 As[2][128][40];
    __shared__ __align__(16) __bf16 Bs[2][128][40];

    int rowc = threadIdx.x >> 2;            // 0..63
    int colc = (threadIdx.x & 3) * 8;       // 0,8,16,24
    const __bf16* gA  = A  + (size_t)(bm0 + rowc)      * K + colc;
    const __bf16* gA2 = A  + (size_t)(bm0 + rowc + 64) * K + colc;
    const __bf16* gB  = Bt + (size_t)(bn0 + rowc)      * K + colc;
    const __bf16* gB2 = Bt + (size_t)(bn0 + rowc + 64) * K + colc;

    auto issue = [&](int buf, int k0) {
        async_b128(gA  + k0, &As[buf][rowc][colc]);
        async_b128(gA2 + k0, &As[buf][rowc + 64][colc]);
        async_b128(gB  + k0, &Bs[buf][rowc][colc]);
        async_b128(gB2 + k0, &Bs[buf][rowc + 64][colc]);
    };

    issue(0, 0);
    int buf = 0;
    for (int k0 = 0; k0 < K; k0 += 32) {
        bool more = (k0 + 32) < K;
        if (more) { issue(buf ^ 1, k0 + 32); wait_async<4>(); }
        else      { wait_async<0>(); }
        __syncthreads();                    // all waves' async tiles visible

        bf16x16 a0 = cmb(ld8(&As[buf][wm * 32 + l16][half * 8]),
                         ld8(&As[buf][wm * 32 + l16][16 + half * 8]));
        bf16x16 a1 = cmb(ld8(&As[buf][wm * 32 + 16 + l16][half * 8]),
                         ld8(&As[buf][wm * 32 + 16 + l16][16 + half * 8]));
        bf16x16 bj[4];
#pragma unroll
        for (int j = 0; j < 4; ++j)
            bj[j] = cmb(ld8(&Bs[buf][wn * 64 + j * 16 + l16][half * 8]),
                        ld8(&Bs[buf][wn * 64 + j * 16 + l16][16 + half * 8]));
#pragma unroll
        for (int j = 0; j < 4; ++j) {
            acc[0][j] = wmma_bf16(a0, bj[j], acc[0][j]);
            acc[1][j] = wmma_bf16(a1, bj[j], acc[1][j]);
        }
        __syncthreads();                    // done reading before next overwrite
        buf ^= 1;
    }
#else
    // fallback: direct-from-global fragments (round-1 path)
    const __bf16* ar0 = A + (size_t)(bm + l16) * K;
    const __bf16* ar1 = ar0 + (size_t)16 * K;
    const __bf16* bc[4];
#pragma unroll
    for (int j = 0; j < 4; ++j) bc[j] = Bt + (size_t)(bn + j * 16 + l16) * K;
    for (int k0 = 0; k0 < K; k0 += 32) {
        if (k0 + 256 < K) {
            __builtin_prefetch(ar0 + k0 + 256, 0, 3);
            __builtin_prefetch(bc[0] + k0 + 256, 0, 3);
        }
        bf16x16 a0 = cmb(ld8(ar0 + k0 + half * 8), ld8(ar0 + k0 + 16 + half * 8));
        bf16x16 a1 = cmb(ld8(ar1 + k0 + half * 8), ld8(ar1 + k0 + 16 + half * 8));
        bf16x16 bj[4];
#pragma unroll
        for (int j = 0; j < 4; ++j)
            bj[j] = cmb(ld8(bc[j] + k0 + half * 8), ld8(bc[j] + k0 + 16 + half * 8));
#pragma unroll
        for (int j = 0; j < 4; ++j) {
            acc[0][j] = wmma_bf16(a0, bj[j], acc[0][j]);
            acc[1][j] = wmma_bf16(a1, bj[j], acc[1][j]);
        }
    }
#endif

#pragma unroll
    for (int i = 0; i < 2; ++i)
#pragma unroll
        for (int j = 0; j < 4; ++j) {
            int col = bn + j * 16 + l16;
            float bv = 0.f;
            if constexpr (MODE != 0) bv = bias[col];
#pragma unroll
            for (int r = 0; r < 8; ++r) {
                int row = bm + i * 16 + half * 8 + r;
                float v = acc[i][j][r];
                size_t idx = (size_t)row * Nn + col;
                if constexpr (MODE == 1)      outb[idx] = (__bf16)gelu_exact(v + bv);
                else if constexpr (MODE == 2) outf[idx] += v + bv;
                else                          outb[idx] = (__bf16)v;
            }
        }
}

// ---------------- Flash attention: x += softmax(Q K^T * 0.125) V ----------------
// grid: B*H*(N/64) blocks, 128 threads (4 waves); each wave owns 16 query rows.
__global__ void flash_attn_kernel(const __bf16* __restrict__ qkv,
                                  const __bf16* __restrict__ Vt,
                                  float* __restrict__ x) {
    int gid = blockIdx.x;
    int mt = gid & 31;              // N/64 = 32 query tiles
    int h  = (gid >> 5) & 15;
    int b  = gid >> 9;
    int wave = threadIdx.x >> 5;
    int lane = threadIdx.x & 31;
    int l16 = lane & 15, half = lane >> 4;
    int qbase = mt * 64 + wave * 16;

    __shared__ __align__(16) __bf16 Ps[4][16][32];

    // Q fragments (16x64), direct from global: contiguous k = 16*half + s, same order as K frags
    const __bf16* qrow = qkv + (size_t)(b * Nz + qbase + l16) * QKVW + h * DHz;
    bf16x16 qf0 = cmb(ld8(qrow + half * 16),      ld8(qrow + half * 16 + 8));
    bf16x16 qf1 = cmb(ld8(qrow + 32 + half * 16), ld8(qrow + 32 + half * 16 + 8));

    f32x8 o[4] = {};
    float m8[8], l8[8];
#pragma unroll
    for (int r = 0; r < 8; ++r) { m8[r] = -1e30f; l8[r] = 0.f; }

    for (int j = 0; j < Nz; j += 32) {
        // K^T B-fragments: per key, contiguous head-dim bytes
        const __bf16* kb0 = qkv + (size_t)(b * Nz + j + l16) * QKVW + Dz + h * DHz;
        const __bf16* kb1 = kb0 + (size_t)16 * QKVW;
        bf16x16 kf00 = cmb(ld8(kb0 + half * 16),      ld8(kb0 + half * 16 + 8));
        bf16x16 kf01 = cmb(ld8(kb0 + 32 + half * 16), ld8(kb0 + 32 + half * 16 + 8));
        bf16x16 kf10 = cmb(ld8(kb1 + half * 16),      ld8(kb1 + half * 16 + 8));
        bf16x16 kf11 = cmb(ld8(kb1 + 32 + half * 16), ld8(kb1 + 32 + half * 16 + 8));

        f32x8 s0 = {}, s1 = {};
        s0 = wmma_bf16(qf0, kf00, s0);  s0 = wmma_bf16(qf1, kf01, s0);
        s1 = wmma_bf16(qf0, kf10, s1);  s1 = wmma_bf16(qf1, kf11, s1);

        // online softmax; C layout: lane = key column, VGPR r = query r (lanes<16) / r+8
#pragma unroll
        for (int r = 0; r < 8; ++r) {
            float v0 = s0[r] * 0.125f, v1 = s1[r] * 0.125f;
            float mx = fmaxf(v0, v1);
            mx = fmaxf(mx, swz_xor<1>(mx));
            mx = fmaxf(mx, swz_xor<2>(mx));
            mx = fmaxf(mx, swz_xor<4>(mx));
            mx = fmaxf(mx, swz_xor<8>(mx));
            float mn = fmaxf(m8[r], mx);
            float sc = __expf(m8[r] - mn);
            m8[r] = mn;
            l8[r] *= sc;
            float p0 = __expf(v0 - mn), p1 = __expf(v1 - mn);
            l8[r] += p0 + p1;
            o[0][r] *= sc; o[1][r] *= sc; o[2][r] *= sc; o[3][r] *= sc;
            int q = r + half * 8;
            Ps[wave][q][l16]      = (__bf16)p0;
            Ps[wave][q][l16 + 16] = (__bf16)p1;
        }
        __syncthreads();

        // P as A-fragment (16 queries x 32 keys) from LDS
        bf16x16 pf = cmb(ld8(&Ps[wave][l16][half * 16]),
                         ld8(&Ps[wave][l16][half * 16 + 8]));
        // V B-fragments from transposed Vt: contiguous keys per head-dim column
#pragma unroll
        for (int t = 0; t < 4; ++t) {
            const __bf16* vb = Vt + ((size_t)((b * Hz + h) * DHz + t * 16 + l16)) * Nz
                                  + j + half * 16;
            bf16x16 vf = cmb(ld8(vb), ld8(vb + 8));
            o[t] = wmma_bf16(pf, vf, o[t]);
        }
        __syncthreads();
    }

    // normalize + residual accumulate into x
#pragma unroll
    for (int r = 0; r < 8; ++r) {
        float l = l8[r];
        l += swz_xor<1>(l);
        l += swz_xor<2>(l);
        l += swz_xor<4>(l);
        l += swz_xor<8>(l);
        float inv = 1.0f / l;
        int row = b * Nz + qbase + r + half * 8;
#pragma unroll
        for (int t = 0; t < 4; ++t) {
            size_t idx = (size_t)row * Dz + h * DHz + t * 16 + l16;
            x[idx] += o[t][r] * inv;
        }
    }
}

// ---------------- host launch ----------------
extern "C" void kernel_launch(void* const* d_in, const int* in_sizes, int n_in,
                              void* d_out, int out_size, void* d_ws, size_t ws_size,
                              hipStream_t stream) {
    (void)in_sizes; (void)n_in; (void)out_size; (void)ws_size;
    const float* x_in  = (const float*)d_in[0];
    const float* ln1_g = (const float*)d_in[1];
    const float* ln1_b = (const float*)d_in[2];
    const float* Wqkv  = (const float*)d_in[3];
    const float* ln2_g = (const float*)d_in[4];
    const float* ln2_b = (const float*)d_in[5];
    const float* W1    = (const float*)d_in[6];
    const float* b1    = (const float*)d_in[7];
    const float* W2    = (const float*)d_in[8];
    const float* b2    = (const float*)d_in[9];
    float* x = (float*)d_out;

    char* ws = (char*)d_ws;
    size_t off = 0;
    __bf16* h_bf   = (__bf16*)(ws + off); off += (size_t)BN * Dz * 2;            // 16.8 MB
    __bf16* qkv_bf = (__bf16*)(ws + off); off += (size_t)BN * QKVW * 2;          // 50.3 MB
    __bf16* Vt     = (__bf16*)(ws + off); off += (size_t)Bz * Hz * DHz * Nz * 2; // 16.8 MB
    __bf16* hf_bf  = (__bf16*)(ws + off); off += (size_t)BN * Fz * 2;            // 67.1 MB
    __bf16* Wtqkv  = (__bf16*)(ws + off); off += (size_t)QKVW * Dz * 2;          // 6.3 MB
    __bf16* Wt1    = (__bf16*)(ws + off); off += (size_t)Fz * Dz * 2;            // 8.4 MB
    __bf16* Wt2    = (__bf16*)(ws + off); off += (size_t)Dz * Fz * 2;            // 8.4 MB

    (void)hipMemcpyAsync(x, x_in, (size_t)BN * Dz * sizeof(float),
                         hipMemcpyDeviceToDevice, stream);

    for (int l = 0; l < Lz; ++l) {
        // ---- attention block ----
        ln_kernel<<<BN, 256, 0, stream>>>(x, ln1_g + l * Dz, ln1_b + l * Dz, h_bf);
        convT_kernel<<<(Dz * QKVW) / 256, 256, 0, stream>>>(
            Wqkv + (size_t)l * Dz * QKVW, Wtqkv, Dz, QKVW);
        gemm_wmma<0><<<dim3(QKVW / 128, BN / 128), 256, 0, stream>>>(
            h_bf, Wtqkv, nullptr, qkv_bf, nullptr, QKVW, Dz);
        transpose_v_kernel<<<(Bz * Hz * DHz * Nz) / 256, 256, 0, stream>>>(qkv_bf, Vt);
        flash_attn_kernel<<<Bz * Hz * (Nz / 64), 128, 0, stream>>>(qkv_bf, Vt, x);

        // ---- MLP block ----
        ln_kernel<<<BN, 256, 0, stream>>>(x, ln2_g + l * Dz, ln2_b + l * Dz, h_bf);
        convT_kernel<<<(Dz * Fz) / 256, 256, 0, stream>>>(
            W1 + (size_t)l * Dz * Fz, Wt1, Dz, Fz);
        gemm_wmma<1><<<dim3(Fz / 128, BN / 128), 256, 0, stream>>>(
            h_bf, Wt1, b1 + l * Fz, hf_bf, nullptr, Fz, Dz);
        convT_kernel<<<(Fz * Dz) / 256, 256, 0, stream>>>(
            W2 + (size_t)l * Fz * Dz, Wt2, Fz, Dz);
        gemm_wmma<2><<<dim3(Dz / 128, BN / 128), 256, 0, stream>>>(
            hf_bf, Wt2, b2 + l * Dz, nullptr, x, Dz, Fz);
    }
}
